// KnowledgeGraphEncoder_14637248544852
// MI455X (gfx1250) — compile-verified, matrix operand order checked by reading, not run on hardware
//
#include <hip/hip_runtime.h>

typedef __attribute__((ext_vector_type(16))) _Float16 v16h;
typedef __attribute__((ext_vector_type(8)))  float    v8f;
typedef __attribute__((ext_vector_type(4)))  unsigned u32x4;
typedef __attribute__((ext_vector_type(8)))  int      i32x8;
typedef __attribute__((ext_vector_type(4)))  int      i32x4;
typedef _Float16 f16;

#define NN 4096
#define ALPHA 0.2f
#define NEG_BIG -9e15f

static __device__ __forceinline__ v8f wmma_f16(v16h a, v16h b, v8f c) {
  return __builtin_amdgcn_wmma_f32_16x16x32_f16(false, a, false, b, (short)0, c, false, false);
}

// LDS byte offset of a __shared__ object (flat LDS pointers keep the LDS
// offset in the low 32 bits; aperture lives in the high bits).
static __device__ __forceinline__ unsigned lds_off(const void* p) {
  return (unsigned)(unsigned long long)(uintptr_t)p;
}

// ---- Tensor Data Mover: DMA a 2D tile (tile_k x tile_rows of f16, row stride
// row_stride_elems) from global memory into LDS. Issued once per calling wave,
// tracked with TENSORcnt. Descriptor layout per CDNA5 ISA section 8.
static __device__ __forceinline__ void tdm_load_2d_f16(
    unsigned lds_byte_off, const void* gaddr,
    unsigned tile_k, unsigned tile_rows, unsigned row_stride_elems) {
  unsigned long long ga = (unsigned long long)gaddr;
  u32x4 g0;
  g0[0] = 1u;                                              // count=1, user mode
  g0[1] = lds_byte_off;                                    // lds_addr
  g0[2] = (unsigned)(ga & 0xffffffffu);                    // global_addr[31:0]
  g0[3] = (unsigned)((ga >> 32) & 0x01ffffffu) | (2u << 30); // addr[56:32] | type=2
  i32x8 g1;
  g1[0] = (int)(1u << 16);                                 // data_size=1 (2 bytes)
  g1[1] = (int)((tile_k & 0xffffu) << 16);                 // tensor_dim0[15:0] (= tile_k)
  g1[2] = (int)(((tile_k >> 16) & 0xffffu) |               // tensor_dim0[31:16]
                ((tile_rows & 0xffffu) << 16));            // tensor_dim1[15:0] (= rows)
  g1[3] = (int)(((tile_rows >> 16) & 0xffffu) |            // tensor_dim1[31:16]
                ((tile_k & 0xffffu) << 16));               // tile_dim0
  g1[4] = (int)(tile_rows & 0xffffu);                      // tile_dim1 (tile_dim2=0)
  g1[5] = (int)row_stride_elems;                           // tensor_dim0_stride[31:0]
  g1[6] = 0;                                               // stride[47:32], dim1_stride lo
  g1[7] = 0;
  i32x4 gz = {0, 0, 0, 0};
#if defined(__clang_major__) && __clang_major__ >= 23
  i32x8 gz8 = {0, 0, 0, 0, 0, 0, 0, 0};
  __builtin_amdgcn_tensor_load_to_lds(g0, g1, gz, gz, gz8, 0);
#else
  __builtin_amdgcn_tensor_load_to_lds(g0, g1, gz, gz, 0);
#endif
}

// ---------------- elementwise f32 -> f16 ----------------
__global__ __launch_bounds__(256) void cvt_k(const float* __restrict__ in,
                                             f16* __restrict__ out, long long n) {
  for (long long i = (long long)blockIdx.x * 256 + threadIdx.x; i < n;
       i += (long long)gridDim.x * 256)
    out[i] = (f16)in[i];
}

// ------------- f32 [R][C] -> f16 transposed [C][R] (per-z strides) -------------
__global__ __launch_bounds__(256) void cvtT_k(const float* __restrict__ in, long long inZ,
                                              f16* __restrict__ out, long long outZ,
                                              int R, int C) {
  in  += inZ  * blockIdx.z;
  out += outZ * blockIdx.z;
  long long total = (long long)R * C;
  for (long long idx = (long long)blockIdx.x * 256 + threadIdx.x; idx < total;
       idx += (long long)gridDim.x * 256) {
    int r = (int)(idx / C), c = (int)(idx % C);
    out[(long long)c * R + r] = (f16)in[idx];
  }
}

// ---------------- generic WMMA GEMM:  C[M,N](f32) = A[M,K](f16) * BT[N,K](f16)^T ----
// block = 256 threads = 8 waves; tile 32(M) x 128(N); K-step 32.
// TDM double-buffered staging: one barrier per K-step.
__global__ __launch_bounds__(256)
void gemm_f16_k(const f16* __restrict__ A, long long aZ,
                const f16* __restrict__ BT, long long bZ,
                float* __restrict__ C, long long cZ,
                int M, int N, int K) {
  __shared__ f16 As[2][32 * 32];
  __shared__ f16 Bs[2][128 * 32];
  A  += aZ * blockIdx.z;
  BT += bZ * blockIdx.z;
  C  += cZ * blockIdx.z;
  const int n0 = blockIdx.x * 128;
  const int m0 = blockIdx.y * 32;
  const int tid  = threadIdx.x;
  const int wave = tid >> 5, lane = tid & 31;
  const int wm = wave & 1, wn = wave >> 1;       // 2 x 4 wave grid
  const int l16 = lane & 15, lhi = lane >> 4;
  const f16* Abase = A  + (long long)m0 * K;
  const f16* Bbase = BT + (long long)n0 * K;

  v8f acc0 = {0.f,0.f,0.f,0.f,0.f,0.f,0.f,0.f};
  v8f acc1 = {0.f,0.f,0.f,0.f,0.f,0.f,0.f,0.f};

  if (wave == 0) {
    tdm_load_2d_f16(lds_off(&As[0][0]), Abase, 32, 32, (unsigned)K);
    tdm_load_2d_f16(lds_off(&Bs[0][0]), Bbase, 32, 128, (unsigned)K);
    __builtin_amdgcn_s_wait_tensorcnt(0);
  }
  __syncthreads();

  int cur = 0;
  for (int kk = 0; kk < K; kk += 32) {
    const bool has_next = (kk + 32 < K);
    if (has_next && wave == 0) {
      tdm_load_2d_f16(lds_off(&As[cur ^ 1][0]), Abase + kk + 32, 32, 32, (unsigned)K);
      tdm_load_2d_f16(lds_off(&Bs[cur ^ 1][0]), Bbase + kk + 32, 32, 128, (unsigned)K);
    }
    union { unsigned u[8]; v16h h; } af, bf0, bf1;
    #pragma unroll
    for (int v = 0; v < 8; ++v) {
      int k0 = ((v < 4) ? 0 : 16) + (lhi ? 8 : 0) + 2 * (v & 3);
      af.u[v]  = *(const unsigned*)&As[cur][(wm * 16 + l16) * 32 + k0];
      bf0.u[v] = *(const unsigned*)&Bs[cur][(wn * 32 + l16) * 32 + k0];
      bf1.u[v] = *(const unsigned*)&Bs[cur][(wn * 32 + 16 + l16) * 32 + k0];
    }
    acc0 = wmma_f16(af.h, bf0.h, acc0);
    acc1 = wmma_f16(af.h, bf1.h, acc1);
    if (has_next && wave == 0) __builtin_amdgcn_s_wait_tensorcnt(0);
    __syncthreads();
    cur ^= 1;
  }
  #pragma unroll
  for (int r = 0; r < 8; ++r) {
    int mr = m0 + wm * 16 + r + 8 * lhi;
    int nc = n0 + wn * 32 + l16;
    C[(long long)mr * N + nc]      = acc0[r];
    C[(long long)mr * N + nc + 16] = acc1[r];
  }
}

// ---------------- attention row scores: s_src/s_dst [H][NN] ----------------
__global__ __launch_bounds__(256)
void scores_k(const float* __restrict__ Wh, const float* __restrict__ a,
              float* __restrict__ s_src, float* __restrict__ s_dst, int O) {
  const int n = blockIdx.x, h = blockIdx.y;
  const float* w  = Wh + ((long long)h * NN + n) * O;
  const float* a1 = a + (long long)h * 2 * O;
  const float* a2 = a1 + O;
  float p1 = 0.f, p2 = 0.f;
  for (int o = threadIdx.x; o < O; o += 256) {
    float v = w[o];
    p1 += v * a1[o];
    p2 += v * a2[o];
  }
  __shared__ float r1[256], r2[256];
  r1[threadIdx.x] = p1; r2[threadIdx.x] = p2;
  __syncthreads();
  for (int s = 128; s > 0; s >>= 1) {
    if (threadIdx.x < s) { r1[threadIdx.x] += r1[threadIdx.x + s];
                           r2[threadIdx.x] += r2[threadIdx.x + s]; }
    __syncthreads();
  }
  if (threadIdx.x == 0) {
    s_src[h * NN + n] = r1[0];
    s_dst[h * NN + n] = r2[0];
  }
}

// ---------------- fused two-pass softmax attention + P @ Wh (WMMA) ----------------
// grid = (NN/16 row-tiles, H). block = 256 (8 waves); wave owns O/8 output cols.
// TDM stages WhT tiles; probability tiles double-buffered; 1 barrier per step.
template <int O>
__global__ __launch_bounds__(256)
void flash_gat_k(const float* __restrict__ s_src, const float* __restrict__ s_dst,
                 const int* __restrict__ adj, const float* __restrict__ ew,
                 const f16* __restrict__ WhT,   // [H][O][NN] f16
                 float* __restrict__ out, long long out_head_stride, int out_row_stride) {
  constexpr int OW = O / 8;
  constexpr int NT = OW / 16;
  __shared__ f16 Pt[2][16 * 32];
  __shared__ f16 Wst[2][O * 32];
  __shared__ float redm[256], redl[256];
  __shared__ float mrow[16], lrow[16], ssrow[16];

  const int head = blockIdx.y;
  const int i0   = blockIdx.x * 16;
  const int tid  = threadIdx.x;
  const float* ss  = s_src + (long long)head * NN;
  const float* sd  = s_dst + (long long)head * NN;
  const f16*   whb = WhT + (long long)head * O * NN;

  if (tid < 16) ssrow[tid] = ss[i0 + tid];

  // ---- phase 1: per-row online (max, sumexp) over all 4096 columns ----
  {
    int row = tid >> 4, g = tid & 15;       // 16 threads scan each row, coalesced in j
    int i = i0 + row;
    float sv = ss[i];
    const int*   arow = adj + (long long)i * NN;
    const float* erow = ew  + (long long)i * NN;
    float m = -3.4e38f, l = 0.f;
    for (int j = g; j < NN; j += 16) {
      float e = sv + sd[j];
      e = e > 0.f ? e : ALPHA * e;
      float v = ((arow[j] != 0) || (j == i)) ? e : NEG_BIG;
      v *= erow[j];
      if (v > m) { l = l * __expf(m - v) + 1.f; m = v; }
      else         l += __expf(v - m);
    }
    redm[row * 16 + g] = m;
    redl[row * 16 + g] = l;
  }
  __syncthreads();
  if (tid < 16) {
    float m = -3.4e38f, l = 0.f;
    for (int g = 0; g < 16; ++g) {
      float m2 = redm[tid * 16 + g], l2 = redl[tid * 16 + g];
      float mn = m > m2 ? m : m2;
      l = l * __expf(m - mn) + l2 * __expf(m2 - mn);
      m = mn;
    }
    mrow[tid] = m; lrow[tid] = l;
  }
  __syncthreads();

  // ---- phase 2: stream column tiles; TDM + P-tile compute overlap WMMA ----
  const int wave = tid >> 5, lane = tid & 31;
  const int l16 = lane & 15, lhi = lane >> 4;

  auto computeP = [&](int j0, f16* Pbuf) {
    #pragma unroll
    for (int e2 = 0; e2 < 2; ++e2) {
      int e = tid + e2 * 256;
      int r = e >> 5, c = e & 31;
      int i = i0 + r, j = j0 + c;
      float x = ssrow[r] + sd[j];
      x = x > 0.f ? x : ALPHA * x;
      float v = ((adj[(long long)i * NN + j] != 0) || (j == i)) ? x : NEG_BIG;
      v *= ew[(long long)i * NN + j];
      Pbuf[e] = (f16)__expf(v - mrow[r]);
    }
  };

  v8f acc[NT];
  #pragma unroll
  for (int t = 0; t < NT; ++t) acc[t] = v8f{0.f,0.f,0.f,0.f,0.f,0.f,0.f,0.f};

  // prologue: DMA tile 0 while computing probability tile 0
  if (wave == 0) tdm_load_2d_f16(lds_off(&Wst[0][0]), whb, 32, O, NN);
  computeP(0, Pt[0]);
  if (wave == 0) __builtin_amdgcn_s_wait_tensorcnt(0);
  __syncthreads();

  int cur = 0;
  for (int j0 = 0; j0 < NN; j0 += 32) {
    const int nxt = j0 + 32;
    if (nxt < NN && wave == 0)
      tdm_load_2d_f16(lds_off(&Wst[cur ^ 1][0]), whb + nxt, 32, O, NN);

    union { unsigned u[8]; v16h h; } af;
    #pragma unroll
    for (int v = 0; v < 8; ++v) {
      int k0 = ((v < 4) ? 0 : 16) + (lhi ? 8 : 0) + 2 * (v & 3);
      af.u[v] = *(const unsigned*)&Pt[cur][l16 * 32 + k0];
    }
    #pragma unroll
    for (int t = 0; t < NT; ++t) {
      int o = wave * OW + t * 16 + l16;
      union { unsigned u[8]; v16h h; } bf;
      #pragma unroll
      for (int v = 0; v < 8; ++v) {
        int k0 = ((v < 4) ? 0 : 16) + (lhi ? 8 : 0) + 2 * (v & 3);
        bf.u[v] = *(const unsigned*)&Wst[cur][o * 32 + k0];
      }
      acc[t] = wmma_f16(af.h, bf.h, acc[t]);
    }

    if (nxt < NN) computeP(nxt, Pt[cur ^ 1]);
    if (nxt < NN && wave == 0) __builtin_amdgcn_s_wait_tensorcnt(0);
    __syncthreads();
    cur ^= 1;
  }

  // epilogue: normalize by row sumexp, write out
  #pragma unroll
  for (int t = 0; t < NT; ++t) {
    #pragma unroll
    for (int r = 0; r < 8; ++r) {
      int Mr = r + 8 * lhi;
      int n  = i0 + Mr;
      float val = acc[t][r] / lrow[Mr];
      out[(long long)head * out_head_stride + (long long)n * out_row_stride +
          wave * OW + t * 16 + l16] = val;
    }
  }
}

// ---------------- layer-0 post: ELU -> +resid+bias -> LN -> ELU ----------------
__global__ __launch_bounds__(256)
void post0_k(float* __restrict__ hp /* in-place -> h1 */, const float* __restrict__ resid,
             const float* __restrict__ bias, const float* __restrict__ g,
             const float* __restrict__ b, f16* __restrict__ h1_16) {
  const int n = blockIdx.x, D = 2048;
  float* row = hp + (long long)n * D;
  const float* rr = resid + (long long)n * D;
  __shared__ float red[256];
  float x[8], s = 0.f;
  #pragma unroll
  for (int t = 0; t < 8; ++t) {
    int k = threadIdx.x + t * 256;
    float v = row[k];
    v = v > 0.f ? v : __expf(v) - 1.f;      // ELU on GAT concat output
    v += rr[k] + bias[k];
    x[t] = v; s += v;
  }
  red[threadIdx.x] = s; __syncthreads();
  for (int st = 128; st > 0; st >>= 1) {
    if (threadIdx.x < st) red[threadIdx.x] += red[threadIdx.x + st];
    __syncthreads();
  }
  float mean = red[0] / (float)D;
  __syncthreads();
  float s2 = 0.f;
  #pragma unroll
  for (int t = 0; t < 8; ++t) { float d = x[t] - mean; s2 += d * d; }
  red[threadIdx.x] = s2; __syncthreads();
  for (int st = 128; st > 0; st >>= 1) {
    if (threadIdx.x < st) red[threadIdx.x] += red[threadIdx.x + st];
    __syncthreads();
  }
  float rs = rsqrtf(red[0] / (float)D + 1e-5f);
  #pragma unroll
  for (int t = 0; t < 8; ++t) {
    int k = threadIdx.x + t * 256;
    float y = (x[t] - mean) * rs * g[k] + b[k];
    y = y > 0.f ? y : __expf(y) - 1.f;      // ELU after LN
    row[k] = y;
    h1_16[(long long)n * D + k] = (f16)y;
  }
}

// ---------------- layer-1 post: mean heads -> +resid+bias -> LN -> d_out ----------------
__global__ __launch_bounds__(256)
void post1_k(const float* __restrict__ hp /* [H][NN][768] */, const float* __restrict__ resid,
             const float* __restrict__ bias, const float* __restrict__ g,
             const float* __restrict__ b, float* __restrict__ out) {
  const int n = blockIdx.x, D = 768;
  __shared__ float red[256];
  float x[3], s = 0.f;
  #pragma unroll
  for (int t = 0; t < 3; ++t) {
    int k = threadIdx.x + t * 256;
    long long idx = (long long)n * D + k;
    float v = 0.25f * (hp[idx] + hp[idx + (long long)NN * D] +
                       hp[idx + 2LL * NN * D] + hp[idx + 3LL * NN * D]);
    v += resid[idx] + bias[k];
    x[t] = v; s += v;
  }
  red[threadIdx.x] = s; __syncthreads();
  for (int st = 128; st > 0; st >>= 1) {
    if (threadIdx.x < st) red[threadIdx.x] += red[threadIdx.x + st];
    __syncthreads();
  }
  float mean = red[0] / (float)D;
  __syncthreads();
  float s2 = 0.f;
  #pragma unroll
  for (int t = 0; t < 3; ++t) { float d = x[t] - mean; s2 += d * d; }
  red[threadIdx.x] = s2; __syncthreads();
  for (int st = 128; st > 0; st >>= 1) {
    if (threadIdx.x < st) red[threadIdx.x] += red[threadIdx.x + st];
    __syncthreads();
  }
  float rs = rsqrtf(red[0] / (float)D + 1e-5f);
  #pragma unroll
  for (int t = 0; t < 3; ++t) {
    int k = threadIdx.x + t * 256;
    out[(long long)n * D + k] = (x[t] - mean) * rs * g[k] + b[k];
  }
}

// =========================================================================
extern "C" void kernel_launch(void* const* d_in, const int* in_sizes, int n_in,
                              void* d_out, int out_size, void* d_ws, size_t ws_size,
                              hipStream_t stream) {
  const float* nf   = (const float*)d_in[0];
  const int*   adj  = (const int*)  d_in[1];
  const float* ew   = (const float*)d_in[2];
  const float* W0   = (const float*)d_in[3];
  const float* a0   = (const float*)d_in[4];
  const float* W1   = (const float*)d_in[5];
  const float* a1   = (const float*)d_in[6];
  const float* rp0w = (const float*)d_in[7];
  const float* rp0b = (const float*)d_in[8];
  const float* rp1w = (const float*)d_in[9];
  const float* rp1b = (const float*)d_in[10];
  const float* ln0g = (const float*)d_in[11];
  const float* ln0b = (const float*)d_in[12];
  const float* ln1g = (const float*)d_in[13];
  const float* ln1b = (const float*)d_in[14];

  char* ws = (char*)d_ws;
  // lifetime-based reuse unions
  const size_t oU1 = 0;                      // h16 (6MB)  | resid1 (12.6MB)
  const size_t oU2 = oU1 + 12582912;         // W0T16+rp0T16 | W1T16+rp1T16 (15.7MB)
  const size_t oU3 = oU2 + 15728640;         // Wh0 (33.6MB) | Wh1 (50.3MB)
  const size_t oU4 = oU3 + 50331648;         // Wh0T16 (16.8) | Wh1T16 (25.2MB)
  const size_t oU5 = oU4 + 25165824;         // s_src/s_dst (128KB)
  const size_t oU6 = oU5 + 131072;           // resid0 (33.6) | hp1 (50.3MB)
  const size_t oU7 = oU6 + 50331648;         // hp0 == h1 (33.6MB, post0 in-place)
  const size_t oU8 = oU7 + 33554432;         // h1_16 (16.8MB)

  f16*   h16    = (f16*)(ws + oU1);
  float* resid1 = (float*)(ws + oU1);
  f16*   W0T16  = (f16*)(ws + oU2);
  f16*   rp0T16 = (f16*)(ws + oU2 + 3145728);
  f16*   W1T16  = (f16*)(ws + oU2);
  f16*   rp1T16 = (f16*)(ws + oU2 + 12582912);
  float* Wh0    = (float*)(ws + oU3);
  float* Wh1    = (float*)(ws + oU3);
  f16*   Wh0T16 = (f16*)(ws + oU4);
  f16*   Wh1T16 = (f16*)(ws + oU4);
  float* s_src  = (float*)(ws + oU5);
  float* s_dst  = (float*)(ws + oU5 + 65536);
  float* resid0 = (float*)(ws + oU6);
  float* hp1    = (float*)(ws + oU6);
  float* hp0    = (float*)(ws + oU7);        // becomes h1 after post0
  f16*   h1_16  = (f16*)(ws + oU8);

  // ---------------- layer 0 ----------------
  cvt_k<<<4096, 256, 0, stream>>>(nf, h16, 4096LL * 768);
  cvtT_k<<<dim3(1024, 1, 4), 256, 0, stream>>>(W0, 768LL * 512, W0T16, 512LL * 768, 768, 512);
  cvtT_k<<<dim3(2048, 1, 1), 256, 0, stream>>>(rp0w, 0, rp0T16, 0, 768, 2048);

  gemm_f16_k<<<dim3(512 / 128, 4096 / 32, 4), 256, 0, stream>>>(
      h16, 0, W0T16, 512LL * 768, Wh0, 4096LL * 512, 4096, 512, 768);
  cvtT_k<<<dim3(2048, 1, 4), 256, 0, stream>>>(Wh0, 4096LL * 512, Wh0T16, 512LL * 4096, 4096, 512);
  scores_k<<<dim3(4096, 4), 256, 0, stream>>>(Wh0, a0, s_src, s_dst, 512);
  gemm_f16_k<<<dim3(2048 / 128, 4096 / 32, 1), 256, 0, stream>>>(
      h16, 0, rp0T16, 0, resid0, 0, 4096, 2048, 768);

  flash_gat_k<512><<<dim3(256, 4), 256, 0, stream>>>(
      s_src, s_dst, adj, ew, Wh0T16, hp0, 512LL /*head stride in concat row*/, 2048);
  post0_k<<<4096, 256, 0, stream>>>(hp0, resid0, rp0b, ln0g, ln0b, h1_16);

  // ---------------- layer 1 ----------------
  cvtT_k<<<dim3(2048, 1, 4), 256, 0, stream>>>(W1, 2048LL * 768, W1T16, 768LL * 2048, 2048, 768);
  cvtT_k<<<dim3(2048, 1, 1), 256, 0, stream>>>(rp1w, 0, rp1T16, 0, 2048, 768);

  gemm_f16_k<<<dim3(768 / 128, 4096 / 32, 4), 256, 0, stream>>>(
      h1_16, 0, W1T16, 768LL * 2048, Wh1, 4096LL * 768, 4096, 768, 2048);
  cvtT_k<<<dim3(2048, 1, 4), 256, 0, stream>>>(Wh1, 4096LL * 768, Wh1T16, 768LL * 4096, 4096, 768);
  scores_k<<<dim3(4096, 4), 256, 0, stream>>>(Wh1, a1, s_src, s_dst, 768);
  gemm_f16_k<<<dim3(768 / 128, 4096 / 32, 1), 256, 0, stream>>>(
      h1_16, 0, rp1T16, 0, resid1, 0, 4096, 768, 2048);

  flash_gat_k<768><<<dim3(256, 4), 256, 0, stream>>>(
      s_src, s_dst, adj, ew, Wh1T16, hp1, 4096LL * 768 /*per-head block*/, 768);
  post1_k<<<4096, 256, 0, stream>>>(hp1, resid1, rp1b, ln1g, ln1b, (float*)d_out);
}
